// RSSM_72121090835048
// MI455X (gfx1250) — matrix operand order repaired and checked by reading, not compiled, not run on hardware
//
#include <hip/hip_runtime.h>

#define B_ 512
#define T_ 512
#define OBS_ 64
#define ACT_ 8
#define H_ 128
#define L_ 32
#define G3 384          // 3*H
#define KG 192          // fused gate K: [act(8) z(32) pad(24) h(128)]
#define KD1 160
#define NWAVE 4
#define NTHREADS 128    // 4 waves (wave32)

typedef _Float16 half_t;
typedef __attribute__((ext_vector_type(16))) _Float16 v16h;
typedef __attribute__((ext_vector_type(8)))  float    v8f;

// LDS layout (bytes)
#define OFF_WG    0         // 384x192 f16 = 147456
#define OFF_PR    147456    // 64x128  f16 = 16384
#define OFF_PO    163840    // 64x192  f16 = 24576
#define OFF_D1    188416    // 128x160 f16 = 40960
#define OFF_D2    229376    // 64x128  f16 = 16384
#define OFF_S     245760    // 16x384  f32 = 24576
#define OFF_GN    270336    // 16x128  f32 = 8192
#define OFF_H     278528    // 16x128  f32 = 8192
#define OFF_AG    286720    // 16x192  f16 = 6144
#define OFF_HZO   292864    // 16x224  f16 = 7168
#define OFF_PP    300032    // 16x128  f32 = 8192
#define OFF_D1H   308224    // 16x128  f16 = 4096
#define OFF_KL    312320    // f32 (+pad)
#define OFF_STGO  312336    // 2 x 16x64 f32 staging (obs)   = 8192
#define OFF_STGA  320528    // 2 x 16x8  f32 staging (act)   = 1024
#define OFF_STGE  321552    // 2 x 16x32 f32 staging (eps)   = 4096
#define LDS_BYTES 325648    // <= 327680 (320KB per WGP)

// ---- CDNA5 async tensor path: ASYNCcnt-tracked global->LDS DMA ----
__device__ __forceinline__ void async_load_b128(unsigned lds_off, const void* gptr) {
  asm volatile("global_load_async_to_lds_b128 %0, %1, off"
               :: "v"(lds_off), "v"(gptr) : "memory");
}
__device__ __forceinline__ void wait_async0() {
#if __has_builtin(__builtin_amdgcn_s_wait_asynccnt)
  __builtin_amdgcn_s_wait_asynccnt(0);
#else
  asm volatile("s_wait_asynccnt 0x0" ::: "memory");
#endif
}

// A fragment: 16xK row-major f16 in LDS, one 16x32 K-tile.
// lane m = lane&15 -> row; lanes 0-15 hold K=kbase..kbase+15, lanes 16-31 hold K=kbase+16..kbase+31.
__device__ __forceinline__ v16h load_a_frag(const half_t* A, int ld, int m, int hi, int kbase) {
  const half_t* p = A + m * ld + kbase + hi * 16;
  v16h a;
#pragma unroll
  for (int e = 0; e < 16; ++e) a[e] = p[e];
  return a;
}
// B fragment built as W^T: W is (N,K) row-major in LDS; lane n = lane&15 -> column of B;
// lanes 0-15 hold K=kbase..kbase+15, lanes 16-31 hold K=kbase+16..kbase+31.
__device__ __forceinline__ v16h load_b_frag(const half_t* W, int ld, int n, int hi, int kbase) {
  const half_t* p = W + n * ld + kbase + hi * 16;
  v16h b;
#pragma unroll
  for (int e = 0; e < 16; ++e) b[e] = p[e];
  return b;
}
__device__ __forceinline__ v8f wmma_f16(v16h a, v16h b, v8f c) {
  return __builtin_amdgcn_wmma_f32_16x16x32_f16(false, a, false, b, (short)0, c, false, false);
}
__device__ __forceinline__ v8f bcast8(float x) {
  v8f c;
#pragma unroll
  for (int v = 0; v < 8; ++v) c[v] = x;
  return c;
}

__global__ void rssm_zero_kls(float* kls) {
  int i = threadIdx.x;
  if (i < T_) kls[i] = 0.f;
}

__global__ __launch_bounds__(NTHREADS) void rssm_kernel(
    const float* __restrict__ obs, const float* __restrict__ actions, const float* __restrict__ eps,
    const float* __restrict__ w_ih, const float* __restrict__ w_hh,
    const float* __restrict__ b_ih, const float* __restrict__ b_hh,
    const float* __restrict__ prior_w, const float* __restrict__ prior_b,
    const float* __restrict__ post_w, const float* __restrict__ post_b,
    const float* __restrict__ dec_w1, const float* __restrict__ dec_b1,
    const float* __restrict__ dec_w2, const float* __restrict__ dec_b2,
    float* __restrict__ out_recons, float* __restrict__ out_kls) {
  extern __shared__ char smem[];
  half_t* sWg  = (half_t*)(smem + OFF_WG);
  half_t* sPr  = (half_t*)(smem + OFF_PR);
  half_t* sPo  = (half_t*)(smem + OFF_PO);
  half_t* sD1  = (half_t*)(smem + OFF_D1);
  half_t* sD2  = (half_t*)(smem + OFF_D2);
  float*  sS   = (float*) (smem + OFF_S);
  float*  sGn  = (float*) (smem + OFF_GN);
  float*  sH   = (float*) (smem + OFF_H);
  half_t* sAg  = (half_t*)(smem + OFF_AG);
  half_t* sHzo = (half_t*)(smem + OFF_HZO);
  float*  sPP  = (float*) (smem + OFF_PP);
  half_t* sD1h = (half_t*)(smem + OFF_D1H);
  float*  sKl  = (float*) (smem + OFF_KL);
  float*  sStgO = (float*)(smem + OFF_STGO);   // [2][16*64]
  float*  sStgA = (float*)(smem + OFF_STGA);   // [2][16*8]
  float*  sStgE = (float*)(smem + OFF_STGE);   // [2][16*32]

  const int tid  = threadIdx.x;
  const int lane = tid & 31;
  const int wave = tid >> 5;
  const int m16  = lane & 15;
  const int hi   = lane >> 4;
  const int bbase = blockIdx.x * 16;

  // ---- prefetch step 0 inputs via async DMA into staging buf 0 ----
  for (int i = tid; i < 256; i += NTHREADS) {          // obs: 256 x 16B
    int m = i >> 4, c = (i & 15) * 4;
    async_load_b128(OFF_STGO + i * 16, &obs[((size_t)(bbase + m) * T_) * OBS_ + c]);
  }
  if (tid < 32) {                                      // act: 32 x 16B
    int m = tid >> 1, c = (tid & 1) * 4;
    async_load_b128(OFF_STGA + tid * 16, &actions[((size_t)(bbase + m) * T_) * ACT_ + c]);
  }
  {                                                    // eps: 128 x 16B
    int m = tid >> 3, c = (tid & 7) * 4;
    async_load_b128(OFF_STGE + tid * 16, &eps[((size_t)(bbase + m) * T_) * L_ + c]);
  }

  // ---- one-time: per-lane bias registers (avoid global reloads inside t-loop) ----
  float bias1[6], biasg[2], biasd1[2];
#pragma unroll
  for (int i = 0; i < 6; ++i) {
    int n = (wave + i * NWAVE) * 16 + m16;
    bias1[i] = b_ih[n] + b_hh[n];
  }
#pragma unroll
  for (int i = 0; i < 2; ++i)
    biasg[i] = b_ih[256 + (wave + i * NWAVE) * 16 + m16];
#pragma unroll
  for (int i = 0; i < 2; ++i)
    biasd1[i] = dec_b1[(wave + i * NWAVE) * 16 + m16];
  const float biasp = prior_b[wave * 16 + m16];
  const float biasq = post_b[wave * 16 + m16];
  const float biasd2 = dec_b2[wave * 16 + m16];

  // ---- one-time: weights -> f16 LDS (overlaps with the async prefetch) ----
  for (int i = tid; i < G3 * KG; i += NTHREADS) {
    int r = i / KG, c = i % KG;
    float v = 0.f;
    if (c < ACT_ + L_) v = w_ih[r * (ACT_ + L_) + c];
    else if (c >= 64)  v = w_hh[r * H_ + (c - 64)];
    sWg[i] = (half_t)v;
  }
  for (int i = tid; i < 2 * L_ * H_;          i += NTHREADS) sPr[i] = (half_t)prior_w[i];
  for (int i = tid; i < 2 * L_ * (H_ + OBS_); i += NTHREADS) sPo[i] = (half_t)post_w[i];
  for (int i = tid; i < 128 * KD1;            i += NTHREADS) sD1[i] = (half_t)dec_w1[i];
  for (int i = tid; i < OBS_ * 128;           i += NTHREADS) sD2[i] = (half_t)dec_w2[i];
  for (int i = tid; i < 16 * H_;              i += NTHREADS) sH[i]  = 0.f;
  for (int i = tid; i < 16 * KG;              i += NTHREADS) sAg[i] = (half_t)0.f;  // h0=z0=0, pad=0

  for (int t = 0; t < T_; ++t) {
    const int cur = t & 1;
    // ---- wait for this step's staged inputs (all waves), then publish ----
    wait_async0();
    __syncthreads();

    // ---- stage 0: convert staged f32 -> f16 working tiles; kick off t+1 DMA ----
    {
      int m = tid >> 3, c = tid & 7;  // exactly 16x8
      sAg[m * KG + c] = (half_t)sStgA[cur * 128 + m * 8 + c];
    }
    for (int i = tid; i < 16 * OBS_; i += NTHREADS)
      sHzo[(i >> 6) * 224 + 160 + (i & 63)] = (half_t)sStgO[cur * 1024 + i];
    if (tid == 0) *sKl = 0.f;

    if (t + 1 < T_) {
      const int nb = (t + 1) & 1;
      for (int i = tid; i < 256; i += NTHREADS) {
        int m = i >> 4, c = (i & 15) * 4;
        async_load_b128(OFF_STGO + nb * 4096 + i * 16,
                        &obs[((size_t)(bbase + m) * T_ + (t + 1)) * OBS_ + c]);
      }
      if (tid < 32) {
        int m = tid >> 1, c = (tid & 1) * 4;
        async_load_b128(OFF_STGA + nb * 512 + tid * 16,
                        &actions[((size_t)(bbase + m) * T_ + (t + 1)) * ACT_ + c]);
      }
      {
        int m = tid >> 3, c = (tid & 7) * 4;
        async_load_b128(OFF_STGE + nb * 2048 + tid * 16,
                        &eps[((size_t)(bbase + m) * T_ + (t + 1)) * L_ + c]);
      }
    }
    __syncthreads();

    // ---- stage 1: fused gate GEMM  s = [act|z|0|h] @ Wg^T + b_ih + b_hh  (16x384) ----
#pragma unroll
    for (int i = 0; i < 6; ++i) {
      int nt = wave + i * NWAVE;
      v8f c = bcast8(bias1[i]);
#pragma unroll
      for (int k = 0; k < KG; k += 32)
        c = wmma_f16(load_a_frag(sAg, KG, m16, hi, k),
                     load_b_frag(sWg, KG, nt * 16 + m16, hi, k), c);
#pragma unroll
      for (int v = 0; v < 8; ++v) sS[(hi * 8 + v) * G3 + nt * 16 + m16] = c[v];
    }
    // gi_n alone (K=64: act+z part) so we can recover gh_n = s_n - gi_n
#pragma unroll
    for (int i = 0; i < 2; ++i) {
      int nt = wave + i * NWAVE;
      v8f c = bcast8(biasg[i]);
#pragma unroll
      for (int k = 0; k < 64; k += 32)
        c = wmma_f16(load_a_frag(sAg, KG, m16, hi, k),
                     load_b_frag(sWg + 256 * KG, KG, nt * 16 + m16, hi, k), c);
#pragma unroll
      for (int v = 0; v < 8; ++v) sGn[(hi * 8 + v) * H_ + nt * 16 + m16] = c[v];
    }
    __syncthreads();

    // ---- stage 2: GRU elementwise (16x128) ----
    {
      int m = tid >> 3, jb = (tid & 7) * 16;
#pragma unroll
      for (int i = 0; i < 16; ++i) {
        int j = jb + i;
        float sr = sS[m * G3 + j];
        float su = sS[m * G3 + 128 + j];
        float sn = sS[m * G3 + 256 + j];
        float gi = sGn[m * H_ + j];
        float hold = sH[m * H_ + j];
        float r = 1.f / (1.f + __expf(-sr));
        float u = 1.f / (1.f + __expf(-su));
        float nn = tanhf(gi + r * (sn - gi));
        float hnew = (1.f - u) * nn + u * hold;
        sH[m * H_ + j] = hnew;
        sHzo[m * 224 + j] = (half_t)hnew;       // for prior/post/dec1
        sAg[m * KG + 64 + j] = (half_t)hnew;    // recurrent input for t+1
      }
    }
    __syncthreads();

    // ---- stage 3: prior (wave -> 1 tile) and posterior (wave -> 1 tile, split-K) ----
    {
      int nt = wave;
      v8f c = bcast8(biasp);
#pragma unroll
      for (int k = 0; k < 128; k += 32)
        c = wmma_f16(load_a_frag(sHzo, 224, m16, hi, k),
                     load_b_frag(sPr, 128, nt * 16 + m16, hi, k), c);
#pragma unroll
      for (int v = 0; v < 8; ++v) sPP[(hi * 8 + v) * 128 + nt * 16 + m16] = c[v];
    }
    {
      int nt = wave;
      v8f c = bcast8(biasq);
#pragma unroll
      for (int k = 0; k < 128; k += 32)        // h part
        c = wmma_f16(load_a_frag(sHzo, 224, m16, hi, k),
                     load_b_frag(sPo, 192, nt * 16 + m16, hi, k), c);
#pragma unroll
      for (int k = 0; k < 64; k += 32)         // obs part (A cols 160..223 <-> W cols 128..191)
        c = wmma_f16(load_a_frag(sHzo + 160, 224, m16, hi, k),
                     load_b_frag(sPo + 128, 192, nt * 16 + m16, hi, k), c);
#pragma unroll
      for (int v = 0; v < 8; ++v) sPP[(hi * 8 + v) * 128 + 64 + nt * 16 + m16] = c[v];
    }
    __syncthreads();

    // ---- stage 4: z sampling + KL partial (16x32); eps read straight from staging ----
    {
      int m = tid >> 3, lb = (tid & 7) * 4;
      float kll = 0.f;
#pragma unroll
      for (int i = 0; i < 4; ++i) {
        int l = lb + i;
        float mup = sPP[m * 128 + l];
        float lvp = sPP[m * 128 + 32 + l];
        float muq = sPP[m * 128 + 64 + l];
        float lvq = sPP[m * 128 + 96 + l];
        float zn = mup + sStgE[cur * 512 + m * L_ + l] * __expf(0.5f * lvp);
        sHzo[m * 224 + 128 + l] = (half_t)zn;   // for dec1 this step
        sAg[m * KG + 8 + l] = (half_t)zn;       // recurrent input for t+1
        float d = muq - mup;
        kll += 1.f + lvq - lvp - (d * d + __expf(lvq)) * __expf(-lvp);
      }
      atomicAdd(sKl, kll);
    }
    __syncthreads();
    if (tid == 0) atomicAdd(&out_kls[t], -0.5f * (*sKl));

    // ---- stage 5: decoder layer 1 + relu -> f16 LDS ----
#pragma unroll
    for (int i = 0; i < 2; ++i) {
      int nt = wave + i * NWAVE;
      v8f c = bcast8(biasd1[i]);
#pragma unroll
      for (int k = 0; k < KD1; k += 32)
        c = wmma_f16(load_a_frag(sHzo, 224, m16, hi, k),
                     load_b_frag(sD1, KD1, nt * 16 + m16, hi, k), c);
#pragma unroll
      for (int v = 0; v < 8; ++v) {
        float x = c[v] > 0.f ? c[v] : 0.f;
        sD1h[(hi * 8 + v) * H_ + nt * 16 + m16] = (half_t)x;
      }
    }
    __syncthreads();

    // ---- stage 6: decoder layer 2 -> global recons ----
    {
      int nt = wave;
      v8f c = bcast8(biasd2);
#pragma unroll
      for (int k = 0; k < 128; k += 32)
        c = wmma_f16(load_a_frag(sD1h, H_, m16, hi, k),
                     load_b_frag(sD2, H_, nt * 16 + m16, hi, k), c);
#pragma unroll
      for (int v = 0; v < 8; ++v) {
        int m = hi * 8 + v;
        out_recons[((size_t)(bbase + m) * T_ + t) * OBS_ + nt * 16 + m16] = c[v];
      }
    }
    __syncthreads();
  }
}

extern "C" void kernel_launch(void* const* d_in, const int* in_sizes, int n_in,
                              void* d_out, int out_size, void* d_ws, size_t ws_size,
                              hipStream_t stream) {
  const float* obs     = (const float*)d_in[0];
  const float* actions = (const float*)d_in[1];
  const float* eps     = (const float*)d_in[2];
  const float* w_ih    = (const float*)d_in[3];
  const float* w_hh    = (const float*)d_in[4];
  const float* b_ih    = (const float*)d_in[5];
  const float* b_hh    = (const float*)d_in[6];
  const float* prior_w = (const float*)d_in[7];
  const float* prior_b = (const float*)d_in[8];
  const float* post_w  = (const float*)d_in[9];
  const float* post_b  = (const float*)d_in[10];
  const float* dec_w1  = (const float*)d_in[11];
  const float* dec_b1  = (const float*)d_in[12];
  const float* dec_w2  = (const float*)d_in[13];
  const float* dec_b2  = (const float*)d_in[14];

  float* recons = (float*)d_out;
  float* kls    = recons + (size_t)B_ * T_ * OBS_;

  hipFuncSetAttribute((const void*)rssm_kernel,
                      hipFuncAttributeMaxDynamicSharedMemorySize, LDS_BYTES);

  hipLaunchKernelGGL(rssm_zero_kls, dim3(1), dim3(512), 0, stream, kls);
  hipLaunchKernelGGL(rssm_kernel, dim3(B_ / 16), dim3(NTHREADS), LDS_BYTES, stream,
                     obs, actions, eps, w_ih, w_hh, b_ih, b_hh,
                     prior_w, prior_b, post_w, post_b,
                     dec_w1, dec_b1, dec_w2, dec_b2, recons, kls);
}
